// SRU_65859028517540
// MI455X (gfx1250) — matrix operand order repaired and checked by reading, not compiled, not run on hardware
//
#include <hip/hip_runtime.h>
#include <hip/hip_bf16.h>
#include <stdint.h>

#define B_ 16
#define T_ 2048
#define D_ 1024
#define H_ 1024
#define EPS_ 1e-5f

typedef __bf16 bf16_t;
typedef __attribute__((ext_vector_type(16))) __bf16 v16bf;
typedef __attribute__((ext_vector_type(8)))  float  v8f;
typedef __attribute__((ext_vector_type(4)))  unsigned int v4u;
typedef __attribute__((ext_vector_type(4)))  unsigned int u32x4;
typedef __attribute__((ext_vector_type(8)))  int i32x8;
typedef __attribute__((ext_vector_type(4)))  int i32x4;

union ABFrag { v16bf v; v4u u[2]; };

// ---------------------------------------------------------------- Phase 0
__global__ void cvt_f32_bf16(const float* __restrict__ in,
                             bf16_t* __restrict__ out, int n) {
  int i = blockIdx.x * blockDim.x + threadIdx.x;
  int stride = gridDim.x * blockDim.x;
  for (; i < n; i += stride) out[i] = (bf16_t)in[i];
}

// ---------------------------------------------------------------- Phase 1
// One workgroup = 16 rows (t) x 3072 cols.  16 waves; wave w owns h-tiles
// [4w, 4w+4) of each of x_tilde / f-gate / r-gate so nd=(1-f)*x_tilde is
// local.  A tile (16x1024 bf16) is DMA'd into LDS by the Tensor Data Mover
// via a 2D D# descriptor; B streams from L2 (weights are 6 MB total).
__global__ __launch_bounds__(512) void sru_gemm_gate(
    const bf16_t* __restrict__ xb,      // [B*T][D] bf16
    const bf16_t* __restrict__ wb,      // [3H][D] bf16
    const float* __restrict__ b_g,
    const float* __restrict__ gamma_g,
    const float* __restrict__ beta_g,
    float* __restrict__ f_ws,           // [B*T][H]
    float* __restrict__ r_ws,           // [B*T][H]
    float* __restrict__ nd_ws)          // [B*T][H]
{
  __shared__ bf16_t As[16 * 1024];      // 32 KB A tile
  __shared__ float  s_sum[16], s_sq[16];

  const int  tid  = threadIdx.x;
  const int  wave = tid >> 5;
  const int  lane = tid & 31;
  const int  half = lane >> 4;          // 0: rows 0-7 / K-lo, 1: rows 8-15 / K-hi
  const int  l16  = lane & 15;
  const long r0   = (long)blockIdx.x * 16;
  const int  th0  = wave * 4;

  if (tid < 16) { s_sum[tid] = 0.f; s_sq[tid] = 0.f; }

  // TDM: one tensor_load_to_lds brings the whole 16x1024 bf16 A tile.
  // D# group0: count=1 | lds_addr | global_addr | type=2 ("image").
  // D# group1: data_size=2B, tensor 1024x16, tile 1024x16, stride 1024.
  if (wave == 0) {
    const uint64_t gaddr = (uint64_t)(uintptr_t)(xb + r0 * D_);
    const uint32_t laddr = (uint32_t)(uintptr_t)(void*)As;
    u32x4 g0;
    g0[0] = 1u;                                            // count=1 (valid)
    g0[1] = laddr;                                         // lds_addr
    g0[2] = (uint32_t)(gaddr & 0xFFFFFFFFu);               // global_addr lo
    g0[3] = (uint32_t)((gaddr >> 32) & 0x01FFFFFFu) | (2u << 30); // hi | type=2
    i32x8 g1;
    g1[0] = (int)(1u << 16);                 // wg_mask=0, data_size=1 (2 bytes)
    g1[1] = (int)(1024u << 16);              // tensor_dim0 = 1024 (low 16)
    g1[2] = (int)(16u << 16);                // td0 hi=0 | tensor_dim1 = 16 (lo)
    g1[3] = (int)(1024u << 16);              // td1 hi=0 | tile_dim0 = 1024
    g1[4] = (int)16;                         // tile_dim1 = 16, tile_dim2 = 0
    g1[5] = (int)1024;                       // tensor_dim0_stride (lo 32)
    g1[6] = 0;                               // td0s hi | td1s lo
    g1[7] = 0;                               // td1s hi
    i32x4 gz = (i32x4)0;
#if __clang_major__ >= 23
    i32x8 gz8 = (i32x8)0;
    __builtin_amdgcn_tensor_load_to_lds(g0, g1, gz, gz, gz8, 0);
#else
    __builtin_amdgcn_tensor_load_to_lds(g0, g1, gz, gz, 0);
#endif
    __builtin_amdgcn_s_wait_tensorcnt(0);
  }
  __syncthreads();

  v8f acc[12];
  #pragma unroll
  for (int j = 0; j < 12; ++j)
    #pragma unroll
    for (int k = 0; k < 8; ++k) acc[j][k] = 0.f;

  for (int kb = 0; kb < D_; kb += 32) {
    // A fragment (16x32 bf16): lanes 0-15 hold K {kb..kb+7, kb+16..kb+23},
    // lanes 16-31 hold K {kb+8..15, kb+24..31}  (ISA 16-bit A layout).
    ABFrag a;
    {
      const bf16_t* ap = As + l16 * D_ + kb + half * 8;
      a.u[0] = *(const v4u*)(ap);
      a.u[1] = *(const v4u*)(ap + 16);
    }
    #pragma unroll
    for (int j = 0; j < 12; ++j) {
      const int m    = j >> 2;                      // 0=x_tilde 1=f 2=r
      const int th   = th0 + (j & 3);
      const int nrow = m * 1024 + th * 16 + l16;    // output column
      // B fragment (32x16): same K layout as A ({kb+8h..+7, kb+16+8h..+7}).
      const bf16_t* bp = wb + (long)nrow * D_ + kb + half * 8;
      ABFrag bfr;
      bfr.u[0] = *(const v4u*)(bp);
      bfr.u[1] = *(const v4u*)(bp + 16);
      acc[j] = __builtin_amdgcn_wmma_f32_16x16x32_bf16(
          false, a.v, false, bfr.v, (short)0, acc[j], false, false);
    }
    if (kb + 32 < D_)
      __builtin_prefetch(wb + (long)(th0 * 16 + l16) * D_ + kb + 32 + half * 16, 0, 3);
  }

  // Bias add + per-row LayerNorm statistics over the 2048 gate columns.
  float rs[8], rq[8];
  #pragma unroll
  for (int i = 0; i < 8; ++i) { rs[i] = 0.f; rq[i] = 0.f; }

  #pragma unroll
  for (int j = 4; j < 12; ++j) {
    const int m    = j >> 2;
    const int th   = th0 + (j & 3);
    const int colp = (m - 1) * 1024 + th * 16 + l16;   // index in [0, 2H)
    const float bias = b_g[colp];
    #pragma unroll
    for (int i = 0; i < 8; ++i) {
      float v = acc[j][i] + bias;
      acc[j][i] = v;
      rs[i] += v;
      rq[i] += v * v;
    }
  }
  // Reduce across the 16 columns held by each lane half, then shared atomics.
  #pragma unroll
  for (int m = 8; m >= 1; m >>= 1)
    #pragma unroll
    for (int i = 0; i < 8; ++i) {
      rs[i] += __shfl_xor(rs[i], m, 32);
      rq[i] += __shfl_xor(rq[i], m, 32);
    }
  if (l16 == 0) {
    #pragma unroll
    for (int i = 0; i < 8; ++i) {
      atomicAdd(&s_sum[i + 8 * half], rs[i]);
      atomicAdd(&s_sq [i + 8 * half], rq[i]);
    }
  }
  __syncthreads();

  float mu[8], rstd[8];
  #pragma unroll
  for (int i = 0; i < 8; ++i) {
    float s = s_sum[i + 8 * half];
    float q = s_sq [i + 8 * half];
    float m_ = s * (1.f / 2048.f);
    float var = q * (1.f / 2048.f) - m_ * m_;
    mu[i] = m_;
    rstd[i] = rsqrtf(var + EPS_);
  }

  // Gate tiles: LN + sigmoid, write f / r; keep f in registers for nd.
  float fvals[4][8];
  #pragma unroll
  for (int j = 4; j < 12; ++j) {
    const int m    = j >> 2;
    const int th   = th0 + (j & 3);
    const int colp = (m - 1) * 1024 + th * 16 + l16;
    const int hcol = th * 16 + l16;
    const float g  = gamma_g[colp];
    const float be = beta_g[colp];
    #pragma unroll
    for (int i = 0; i < 8; ++i) {
      const long row = r0 + i + 8 * half;
      float z = (acc[j][i] - mu[i]) * rstd[i] * g + be;
      float s = 1.f / (1.f + __expf(-z));
      const long idx = row * H_ + hcol;
      if (m == 1) { f_ws[idx] = s; fvals[j - 4][i] = s; }
      else        { r_ws[idx] = s; }
    }
  }
  // x_tilde tiles: nd = (1 - f) * x_tilde
  #pragma unroll
  for (int j = 0; j < 4; ++j) {
    const int th   = th0 + j;
    const int hcol = th * 16 + l16;
    #pragma unroll
    for (int i = 0; i < 8; ++i) {
      const long row = r0 + i + 8 * half;
      nd_ws[row * H_ + hcol] = (1.f - fvals[j][i]) * acc[j][i];
    }
  }
}

// ---------------------------------------------------------------- Phase 2
// Sequential scan over t.  c_t overwrites nd in place (each slot is read
// exactly once before being written).  16384 independent (b,h) chains.
__global__ __launch_bounds__(256) void sru_scan(
    const float* __restrict__ f_ws,
    float* __restrict__ ndc,            // in: nd, out: all_c
    const float* __restrict__ c0,
    float* __restrict__ c_final)
{
  const int b = blockIdx.x >> 2;                      // H/256 = 4 blocks / batch
  const int h = ((blockIdx.x & 3) << 8) + threadIdx.x;
  float c = c0[b * H_ + h];
  size_t idx = (size_t)b * T_ * H_ + h;
  for (int t = 0; t < T_; ++t, idx += H_) {
    __builtin_prefetch(f_ws + idx + 8 * H_, 0, 0);
    __builtin_prefetch(ndc  + idx + 8 * H_, 0, 0);
    float fv = f_ws[idx];
    float nd = ndc[idx];
    c = fmaf(fv, c, nd);
    ndc[idx] = c;
  }
  c_final[b * H_ + h] = c;
}

// ---------------------------------------------------------------- Phase 3
// Row LayerNorm over H=1024 + sigmoid + blend with x via r.
__global__ __launch_bounds__(256) void sru_out(
    const float* __restrict__ allc,
    const float* __restrict__ r_ws,
    const float* __restrict__ x,
    const float* __restrict__ gamma_a,
    const float* __restrict__ beta_a,
    float* __restrict__ hout)
{
  __shared__ float ssum[8], ssq[8];
  const size_t row = blockIdx.x;
  const int tid = threadIdx.x;

  const float4 cv = ((const float4*)(allc + row * H_))[tid];
  float s = cv.x + cv.y + cv.z + cv.w;
  float q = cv.x * cv.x + cv.y * cv.y + cv.z * cv.z + cv.w * cv.w;
  #pragma unroll
  for (int m = 16; m >= 1; m >>= 1) {
    s += __shfl_xor(s, m, 32);
    q += __shfl_xor(q, m, 32);
  }
  if ((tid & 31) == 0) { ssum[tid >> 5] = s; ssq[tid >> 5] = q; }
  __syncthreads();
  s = 0.f; q = 0.f;
  #pragma unroll
  for (int w = 0; w < 8; ++w) { s += ssum[w]; q += ssq[w]; }
  const float mu   = s * (1.f / 1024.f);
  const float rstd = rsqrtf(q * (1.f / 1024.f) - mu * mu + EPS_);

  const float4 rv = ((const float4*)(r_ws + row * H_))[tid];
  const float4 xv = ((const float4*)(x + row * (size_t)D_))[tid];
  const float4 gv = ((const float4*)gamma_a)[tid];
  const float4 bv = ((const float4*)beta_a)[tid];

  float4 o;
  o.x = rv.x * (1.f / (1.f + __expf(-((cv.x - mu) * rstd * gv.x + bv.x)))) + (1.f - rv.x) * xv.x;
  o.y = rv.y * (1.f / (1.f + __expf(-((cv.y - mu) * rstd * gv.y + bv.y)))) + (1.f - rv.y) * xv.y;
  o.z = rv.z * (1.f / (1.f + __expf(-((cv.z - mu) * rstd * gv.z + bv.z)))) + (1.f - rv.z) * xv.z;
  o.w = rv.w * (1.f / (1.f + __expf(-((cv.w - mu) * rstd * gv.w + bv.w)))) + (1.f - rv.w) * xv.w;
  ((float4*)(hout + row * H_))[tid] = o;
}

// ---------------------------------------------------------------- launch
extern "C" void kernel_launch(void* const* d_in, const int* in_sizes, int n_in,
                              void* d_out, int out_size, void* d_ws, size_t ws_size,
                              hipStream_t stream) {
  (void)in_sizes; (void)n_in; (void)out_size; (void)ws_size;
  const float* x       = (const float*)d_in[0];
  const float* c0      = (const float*)d_in[1];
  const float* W_lt    = (const float*)d_in[2];
  const float* W_g     = (const float*)d_in[3];
  const float* b_g     = (const float*)d_in[4];
  const float* gamma_g = (const float*)d_in[5];
  const float* beta_g  = (const float*)d_in[6];
  const float* gamma_a = (const float*)d_in[7];
  const float* beta_a  = (const float*)d_in[8];

  float* h_out   = (float*)d_out;                       // [B*T*H]
  float* c_final = h_out + (size_t)B_ * T_ * H_;        // [B*H]

  char* ws = (char*)d_ws;
  bf16_t* xb  = (bf16_t*)ws;                                            // 64 MB
  bf16_t* wb  = (bf16_t*)(ws + (size_t)B_ * T_ * D_ * 2);               // 6 MB
  float*  f_ws = (float*)(ws + (size_t)B_ * T_ * D_ * 2 + (size_t)3 * H_ * D_ * 2);
  float*  r_ws = f_ws + (size_t)B_ * T_ * H_;
  float*  nd_ws = r_ws + (size_t)B_ * T_ * H_;

  cvt_f32_bf16<<<8192, 256, 0, stream>>>(x, xb, B_ * T_ * D_);
  cvt_f32_bf16<<<2048, 256, 0, stream>>>(W_lt, wb, H_ * D_);
  cvt_f32_bf16<<<4096, 256, 0, stream>>>(W_g, wb + (size_t)H_ * D_, 2 * H_ * D_);

  sru_gemm_gate<<<(B_ * T_) / 16, 512, 0, stream>>>(
      xb, wb, b_g, gamma_g, beta_g, f_ws, r_ws, nd_ws);

  sru_scan<<<B_ * (H_ / 256), 256, 0, stream>>>(f_ws, nd_ws, c0, c_final);

  sru_out<<<B_ * T_, 256, 0, stream>>>(nd_ws, r_ws, x, gamma_a, beta_a, h_out);
}
